// Wbw_Attenion_35381940584535
// MI455X (gfx1250) — compile-verified, matrix operand order checked by reading, not run on hardware
//
#include <hip/hip_runtime.h>
#include <hip/hip_bf16.h>

// ---------------- constants ----------------
constexpr int B_ = 64;     // batch
constexpr int S_ = 384;    // seq len
constexpr int E_ = 64;     // hidden
constexpr int D_ = 50;     // embedding dim (padded to 64 for WMMA K)
constexpr int G_ = 256;    // 4*E gates
constexpr int AP_ = E_ + 1; // padded row stride (floats) for attention LDS staging

// ---------------- WMMA / TDM types ----------------
typedef __attribute__((ext_vector_type(16))) __bf16          v16bf;
typedef __attribute__((ext_vector_type(16))) unsigned short  v16u;
typedef __attribute__((ext_vector_type(8)))  float           v8f;
typedef __attribute__((ext_vector_type(4)))  unsigned int    v4u;
typedef __attribute__((ext_vector_type(8)))  int             v8i;
typedef __attribute__((ext_vector_type(4)))  int             v4i;

__device__ inline unsigned short f2bf(float f) {
  unsigned int u = __builtin_bit_cast(unsigned int, f);
  unsigned int r = u + 0x7FFFu + ((u >> 16) & 1u);   // round-to-nearest-even
  return (unsigned short)(r >> 16);
}

// Fragment loader for row-major [.., K]: elements i=0..7 at k=kbase+i, i=8..15 at
// k=kbase+8+i. Matches CDNA5 16-bit A/B VGPR layout (ISA 7.12.2) with
// kbase = ktile*32 + (lane>>4)*8.
__device__ inline v16u load16_rowK(const unsigned short* rowp, int kbase) {
  v16u r;
#pragma unroll
  for (int i = 0; i < 8; i++) r[i] = rowp[kbase + i];
#pragma unroll
  for (int i = 0; i < 8; i++) r[8 + i] = rowp[kbase + 16 + i];
  return r;
}

__device__ inline v16u load16_f32K(const float* rowp, int kbase) {
  v16u r;
#pragma unroll
  for (int i = 0; i < 8; i++) r[i] = f2bf(rowp[kbase + i]);
#pragma unroll
  for (int i = 0; i < 8; i++) r[8 + i] = f2bf(rowp[kbase + 16 + i]);
  return r;
}

__device__ inline v8f wmma_bf16(v16u a, v16u b, v8f c) {
  return __builtin_amdgcn_wmma_f32_16x16x32_bf16(
      false, __builtin_bit_cast(v16bf, a),
      false, __builtin_bit_cast(v16bf, b),
      (short)0, c, false, false);
}

__device__ inline v8f splat8(float x) {
  v8f c;
#pragma unroll
  for (int v = 0; v < 8; v++) c[v] = x;
  return c;
}

// ---------------- fast transcendentals (v_exp_f32 / v_rcp_f32) ----------------
__device__ inline float fast_tanh(float x) {
  float e = __builtin_amdgcn_exp2f(x * 2.88539008177793f);     // exp(2x)
  return 1.0f - 2.0f * __builtin_amdgcn_rcpf(e + 1.0f);
}
__device__ inline float fast_sig(float x) {
  float e = __builtin_amdgcn_exp2f(-x * 1.4426950408889634f);  // exp(-x)
  return __builtin_amdgcn_rcpf(1.0f + e);
}

// ---------------- TDM: async-copy one 16x256 f32 tile (row-contiguous) to LDS ----
// D# built per ISA 8.3/8.4: 2D tile, tile_dim0=256 elem, tile_dim1=16 rows,
// data_size=4B, tensor_dim0_stride=256; pad_enable with interval=256 DW and
// amount=4 DW so TDM writes LDS rows at stride 260 floats (bank-conflict-free).
__device__ inline void tdm_load_tile_16x256(const float* gsrc, unsigned lds_off) {
  unsigned long long ga = (unsigned long long)(const void*)gsrc;
  v4u g0;
  g0[0] = 1u;                                                  // count=1, user D#
  g0[1] = lds_off;                                             // lds_addr (bytes)
  g0[2] = (unsigned)(ga & 0xFFFFFFFFu);                        // global_addr[31:0]
  g0[3] = (unsigned)((ga >> 32) & 0x01FFFFFFu) | 0x80000000u;  // addr[56:32] | type=2
  v8i g1;
  g1[0] = (int)((2u << 16) | (1u << 20) | (7u << 22) | (3u << 25)); // 4B, pad en, 256DW/4DW
  g1[1] = (int)(256u << 16);                                   // tensor_dim0 = 256
  g1[2] = (int)0xFFFF0000u;                                    // tensor_dim1 lo
  g1[3] = (int)((256u << 16) | 0x00FFu);                       // tensor_dim1 hi | tile_dim0=256
  g1[4] = 16;                                                  // tile_dim1=16, tile_dim2=0
  g1[5] = 256;                                                 // tensor_dim0_stride=256
  g1[6] = 0;
  g1[7] = 0;
  v4i z4 = {0, 0, 0, 0};
  v8i z8 = {0, 0, 0, 0, 0, 0, 0, 0};
  // 6-arg form (this toolchain): (g0, g1, g2, g3, g4, cpol)
  __builtin_amdgcn_tensor_load_to_lds(g0, g1, z4, z4, z8, 0);
}

// ---------------- prep kernels ----------------
__global__ void k_cvt_pad(const float* __restrict__ src, unsigned short* __restrict__ dst,
                          int rows, int sK, int dK) {
  int n = rows * dK;
  for (int i = blockIdx.x * blockDim.x + threadIdx.x; i < n; i += gridDim.x * blockDim.x) {
    int r = i / dK, k = i - r * dK;
    float v = (k < sK) ? src[r * sK + k] : 0.0f;
    dst[i] = f2bf(v);
  }
}
__global__ void k_cvt_T(const float* __restrict__ src, unsigned short* __restrict__ dst,
                        int K, int N) {
  int n = K * N;
  for (int i = blockIdx.x * blockDim.x + threadIdx.x; i < n; i += gridDim.x * blockDim.x) {
    int nn = i / K, k = i - nn * K;
    dst[i] = f2bf(src[k * N + nn]);
  }
}
__global__ void k_bias(const float* __restrict__ a, const float* __restrict__ b,
                       float* __restrict__ o, int n) {
  for (int i = blockIdx.x * blockDim.x + threadIdx.x; i < n; i += gridDim.x * blockDim.x)
    o[i] = a[i] + b[i];
}

// ---------------- input projection (WMMA) ----------------
__global__ void k_xproj(const int* __restrict__ sent1, const int* __restrict__ sent2,
                        const float* __restrict__ emb,
                        const unsigned short* __restrict__ Wb1, const unsigned short* __restrict__ Wb2,
                        const float* __restrict__ bias1, const float* __restrict__ bias2,
                        float* __restrict__ xp1, float* __restrict__ xp2) {
  const int* sent          = blockIdx.y ? sent2 : sent1;
  const unsigned short* Wb = blockIdx.y ? Wb2 : Wb1;
  const float* bias        = blockIdx.y ? bias2 : bias1;
  float* xp                = blockIdx.y ? xp2 : xp1;

  int wave = threadIdx.x >> 5, lane = threadIdx.x & 31;
  int mrow = lane & 15, hilo = lane >> 4;
  int rowtile = blockIdx.x * 8 + wave;          // 0..1535
  int arow = rowtile * 16 + mrow;               // row = s*64 + b
  int s = arow >> 6, b = arow & 63;
  int tok = sent[b * S_ + s];
  const float* erow = emb + (long)tok * D_;

  v16u a[2];
#pragma unroll
  for (int kt = 0; kt < 2; ++kt) {
    int kbase = kt * 32 + hilo * 8;
    v16u r;
#pragma unroll
    for (int i = 0; i < 8; i++) { int k = kbase + i;      r[i]     = (k < D_) ? f2bf(erow[k]) : (unsigned short)0; }
#pragma unroll
    for (int i = 0; i < 8; i++) { int k = kbase + 16 + i; r[8 + i] = (k < D_) ? f2bf(erow[k]) : (unsigned short)0; }
    a[kt] = r;
  }

  for (int nt = 0; nt < 16; ++nt) {
    int n = nt * 16 + mrow;
    const unsigned short* brow = Wb + n * 64;
    v16u b0 = load16_rowK(brow, hilo * 8);
    v16u b1 = load16_rowK(brow, 32 + hilo * 8);
    v8f c = splat8(bias[n]);
    c = wmma_bf16(a[0], b0, c);
    c = wmma_bf16(a[1], b1, c);
#pragma unroll
    for (int v = 0; v < 8; v++)
      xp[(long)(rowtile * 16 + v + 8 * hilo) * G_ + n] = c[v];
  }
}

// ---------------- LSTM scan: TDM double-buffered xp tiles + per-step WMMA ----------------
__global__ void k_lstm(const float* __restrict__ xp, const unsigned short* __restrict__ Whhb,
                       float* __restrict__ hout, float* __restrict__ cout_,
                       const int* __restrict__ glen, const float* __restrict__ hsrc,
                       const float* __restrict__ csrc) {
  __shared__ float xp_s[2][16][260];            // TDM-padded rows (260 = 256 + 4 DW pad)
  __shared__ float gates[16][260];
  __shared__ unsigned short h_lds[16][72];

  int tid = threadIdx.x;
  int wave = tid >> 5, lane = tid & 31, mrow = lane & 15, hilo = lane >> 4;
  int btile = blockIdx.x;

  float creg[4];
#pragma unroll
  for (int q = 0; q < 4; q++) {
    int idx = tid + 256 * q, bl = idx >> 6, j = idx & 63;
    int b = btile * 16 + bl;
    float h0 = 0.f, c0 = 0.f;
    if (glen) {                                  // per-element gather at s1_len (B,1,E)
      int si = glen[b * E_ + j];
      si = si < 0 ? 0 : (si >= S_ ? S_ - 1 : si);
      h0 = hsrc[((long)b * S_ + si) * E_ + j];
      c0 = csrc[((long)b * S_ + si) * E_ + j];
    }
    creg[q] = c0;
    h_lds[bl][j] = f2bf(h0);
  }

  v16u bfrag[2][2];
#pragma unroll
  for (int nn = 0; nn < 2; nn++) {
    int n = (2 * wave + nn) * 16 + mrow;
    const unsigned short* brow = Whhb + n * 64;
    bfrag[nn][0] = load16_rowK(brow, hilo * 8);
    bfrag[nn][1] = load16_rowK(brow, 32 + hilo * 8);
  }

  // prologue: async-stage step 0 tile
  if (tid < 32) {
    tdm_load_tile_16x256(xp + ((long)0 * B_ + btile * 16) * G_,
                         (unsigned)(unsigned long long)(const void*)&xp_s[0][0][0]);
  }

  for (int s = 0; s < S_; s++) {
    int cur = s & 1;
    if (tid < 32) __builtin_amdgcn_s_wait_tensorcnt(0);  // tile for step s complete
    __syncthreads();                                     // tile + h_lds visible to all
    if (tid < 32 && s + 1 < S_) {                        // prefetch next tile (other buffer)
      tdm_load_tile_16x256(xp + ((long)(s + 1) * B_ + btile * 16) * G_,
                           (unsigned)(unsigned long long)(const void*)&xp_s[cur ^ 1][0][0]);
    }

    v16u a0 = load16_rowK(&h_lds[mrow][0], hilo * 8);
    v16u a1 = load16_rowK(&h_lds[mrow][0], 32 + hilo * 8);
#pragma unroll
    for (int nn = 0; nn < 2; nn++) {
      int n = (2 * wave + nn) * 16 + mrow;
      v8f c;
#pragma unroll
      for (int v = 0; v < 8; v++)                        // C init = xproj tile (LDS, has bias)
        c[v] = xp_s[cur][v + 8 * hilo][n];
      c = wmma_bf16(a0, bfrag[nn][0], c);
      c = wmma_bf16(a1, bfrag[nn][1], c);
#pragma unroll
      for (int v = 0; v < 8; v++) gates[v + 8 * hilo][n] = c[v];
    }
    __syncthreads();
#pragma unroll
    for (int q = 0; q < 4; q++) {
      int idx = tid + 256 * q, bl = idx >> 6, j = idx & 63;
      int b = btile * 16 + bl;
      float ig = gates[bl][j], fg = gates[bl][64 + j];
      float gg = gates[bl][128 + j], og = gates[bl][192 + j];
      float c_ = fast_sig(fg) * creg[q] + fast_sig(ig) * fast_tanh(gg);
      float h_ = fast_sig(og) * fast_tanh(c_);
      creg[q] = c_;
      hout[((long)b * S_ + s) * E_ + j] = h_;
      if (cout_) cout_[((long)b * S_ + s) * E_ + j] = c_;
      h_lds[bl][j] = f2bf(h_);
    }
  }
}

// ---------------- base = h1@wy + h2@wh (WMMA) ----------------
__global__ void k_base(const float* __restrict__ h1, const float* __restrict__ h2,
                       const unsigned short* __restrict__ wyT, const unsigned short* __restrict__ whT,
                       float* __restrict__ baseo) {
  int wave = threadIdx.x >> 5, lane = threadIdx.x & 31;
  int mrow = lane & 15, hilo = lane >> 4;
  int rowtile = blockIdx.x * 8 + wave;
  int arow = rowtile * 16 + mrow;
  const float* p1 = h1 + (long)arow * E_;
  const float* p2 = h2 + (long)arow * E_;
  v16u a1k0 = load16_f32K(p1, hilo * 8), a1k1 = load16_f32K(p1, 32 + hilo * 8);
  v16u a2k0 = load16_f32K(p2, hilo * 8), a2k1 = load16_f32K(p2, 32 + hilo * 8);
#pragma unroll
  for (int nt = 0; nt < 4; nt++) {
    int n = nt * 16 + mrow;
    v8f c = splat8(0.0f);
    c = wmma_bf16(a1k0, load16_rowK(wyT + n * 64, hilo * 8), c);
    c = wmma_bf16(a1k1, load16_rowK(wyT + n * 64, 32 + hilo * 8), c);
    c = wmma_bf16(a2k0, load16_rowK(whT + n * 64, hilo * 8), c);
    c = wmma_bf16(a2k1, load16_rowK(whT + n * 64, 32 + hilo * 8), c);
#pragma unroll
    for (int v = 0; v < 8; v++)
      baseo[(long)(rowtile * 16 + v + 8 * hilo) * E_ + n] = c[v];
  }
}

// ---------------- attention scan: per-batch block, base/h staged in 195KB LDS ----------------
__global__ void k_attn(const float* __restrict__ baseb_, const float* __restrict__ h1,
                       const float* __restrict__ wr, const float* __restrict__ wt,
                       const float* __restrict__ w, const float* __restrict__ s1s,
                       const int* __restrict__ s2len, float* __restrict__ rn) {
  extern __shared__ float dynsm[];              // [S_][AP_] base + [S_][AP_] h
  float* base_s = dynsm;
  float* h_s    = dynsm + S_ * AP_;
  __shared__ float r_s[E_], rw_s[E_], twt_s[E_];
  __shared__ float score[S_];
  __shared__ float mask_s[S_];
  __shared__ float red[256];

  int b = blockIdx.x, tid = threadIdx.x;
  const float* baseb = baseb_ + (long)b * S_ * E_;
  const float* hb    = h1 + (long)b * S_ * E_;

  // one-time staging (coalesced global -> padded LDS rows)
  for (int idx = tid; idx < S_ * E_; idx += 256) {
    int s2 = idx >> 6, e = idx & 63;
    base_s[s2 * AP_ + e] = baseb[idx];
    h_s[s2 * AP_ + e]    = hb[idx];
  }
  for (int s2 = tid; s2 < S_; s2 += 256) mask_s[s2] = s1s[b * S_ + s2];
  if (tid < E_) r_s[tid] = 0.f;
  __syncthreads();

  for (int t = 0; t < S_; t++) {
    if (tid < 128) {
      int e = tid & 63;
      const float* wm = (tid < 64) ? wr : wt;
      float acc = 0.f;
#pragma unroll 8
      for (int k = 0; k < E_; k++) acc += r_s[k] * wm[k * E_ + e];
      if (tid < 64) rw_s[e] = acc; else twt_s[e] = fast_tanh(acc);
    }
    __syncthreads();

    float localmax = -3.0e38f;
    for (int s2 = tid; s2 < S_; s2 += 256) {
      const float* br = base_s + s2 * AP_;
      float acc = 0.f;
#pragma unroll 8
      for (int e = 0; e < E_; e++) acc += fast_tanh(br[e] + rw_s[e]) * w[e];
      float msk = mask_s[s2];
      float sc = msk * acc - (1.f - msk) * 1e12f;
      score[s2] = sc;
      localmax = fmaxf(localmax, sc);
    }
    red[tid] = localmax;
    __syncthreads();
    for (int off = 128; off > 0; off >>= 1) {
      if (tid < off) red[tid] = fmaxf(red[tid], red[tid + off]);
      __syncthreads();
    }
    float mx = red[0];
    __syncthreads();

    float lsum = 0.f;
    for (int s2 = tid; s2 < S_; s2 += 256) {
      float p = __builtin_amdgcn_exp2f((score[s2] - mx) * 1.4426950408889634f);
      score[s2] = p;
      lsum += p;
    }
    red[tid] = lsum;
    __syncthreads();
    for (int off = 128; off > 0; off >>= 1) {
      if (tid < off) red[tid] += red[tid + off];
      __syncthreads();
    }
    float inv = __builtin_amdgcn_rcpf(red[0]);
    __syncthreads();

    {
      int e = tid & 63, sg = tid >> 6;
      float acc = 0.f;
      for (int s2 = sg * 96; s2 < sg * 96 + 96; s2++) acc += h_s[s2 * AP_ + e] * score[s2];
      red[tid] = acc;
    }
    __syncthreads();
    if (tid < 64) {
      float rnew = (red[tid] + red[tid + 64] + red[tid + 128] + red[tid + 192]) * inv + twt_s[tid];
      r_s[tid] = rnew;
      if (s2len[b * E_ + tid] == t) rn[b * E_ + tid] = rnew;  // inline Rt gather
    }
    __syncthreads();
  }
}

// ---------------- final MLP head ----------------
__global__ void k_final(const float* __restrict__ rn, const float* __restrict__ out2,
                        const int* __restrict__ s2len,
                        const float* __restrict__ wp, const float* __restrict__ wx,
                        const float* __restrict__ l1W, const float* __restrict__ l1b,
                        const float* __restrict__ lW, const float* __restrict__ lb,
                        float* __restrict__ out) {
  __shared__ float sm[12288];
  int tid = threadIdx.x;
#pragma unroll 4
  for (int q = 0; q < 16; q++) {
    int idx = tid + 256 * q, b = idx >> 6, e = idx & 63;
    int si = s2len[b * E_ + e];
    si = si < 0 ? 0 : (si >= S_ ? S_ - 1 : si);
    sm[idx] = rn[b * E_ + e];
    sm[4096 + idx] = out2[((long)b * S_ + si) * E_ + e];
  }
  __syncthreads();
#pragma unroll 4
  for (int q = 0; q < 16; q++) {
    int idx = tid + 256 * q, b = idx >> 6, e = idx & 63;
    float acc = 0.f;
#pragma unroll 8
    for (int k = 0; k < E_; k++)
      acc += sm[b * 64 + k] * wp[k * E_ + e] + sm[4096 + b * 64 + k] * wx[k * E_ + e];
    sm[8192 + idx] = fast_tanh(acc);
  }
  __syncthreads();
#pragma unroll 4
  for (int q = 0; q < 32; q++) {
    int idx = tid + 256 * q, b = idx >> 7, n = idx & 127;
    float acc = l1b[n];
#pragma unroll 8
    for (int k = 0; k < E_; k++) acc += sm[8192 + b * 64 + k] * l1W[n * E_ + k];
    sm[idx] = fast_tanh(acc);
  }
  __syncthreads();
  {
    int b = tid >> 2, o = tid & 3;
    float acc = lb[o];
#pragma unroll 8
    for (int k = 0; k < 128; k++) acc += sm[b * 128 + k] * lW[o * 128 + k];
    out[b * 4 + o] = acc;
  }
}

// ---------------- launch ----------------
extern "C" void kernel_launch(void* const* d_in, const int* in_sizes, int n_in,
                              void* d_out, int out_size, void* d_ws, size_t ws_size,
                              hipStream_t stream) {
  const int*   sentence1 = (const int*)d_in[0];
  const int*   sentence2 = (const int*)d_in[1];
  const int*   s1_len    = (const int*)d_in[2];
  const int*   s2_len    = (const int*)d_in[3];
  const float* s1_s      = (const float*)d_in[4];
  const float* emb  = (const float*)d_in[6];
  const float* Wih1 = (const float*)d_in[7];
  const float* Whh1 = (const float*)d_in[8];
  const float* bih1 = (const float*)d_in[9];
  const float* bhh1 = (const float*)d_in[10];
  const float* Wih2 = (const float*)d_in[11];
  const float* Whh2 = (const float*)d_in[12];
  const float* bih2 = (const float*)d_in[13];
  const float* bhh2 = (const float*)d_in[14];
  const float* wy   = (const float*)d_in[15];
  const float* wh   = (const float*)d_in[16];
  const float* w    = (const float*)d_in[17];
  const float* wp   = (const float*)d_in[18];
  const float* wx   = (const float*)d_in[19];
  const float* wr   = (const float*)d_in[20];
  const float* wt   = (const float*)d_in[21];
  const float* l1W  = (const float*)d_in[22];
  const float* l1b  = (const float*)d_in[23];
  const float* lW   = (const float*)d_in[24];
  const float* lb   = (const float*)d_in[25];
  float* out = (float*)d_out;

  char* ws = (char*)d_ws;
  size_t off = 0;
  auto take = [&](size_t bytes) -> char* {
    char* p = ws + off;
    off = (off + bytes + 255) & ~(size_t)255;
    return p;
  };
  unsigned short* wih1b = (unsigned short*)take((size_t)G_ * 64 * 2);
  unsigned short* wih2b = (unsigned short*)take((size_t)G_ * 64 * 2);
  unsigned short* whh1b = (unsigned short*)take((size_t)G_ * 64 * 2);
  unsigned short* whh2b = (unsigned short*)take((size_t)G_ * 64 * 2);
  unsigned short* wyT   = (unsigned short*)take((size_t)E_ * E_ * 2);
  unsigned short* whT   = (unsigned short*)take((size_t)E_ * E_ * 2);
  float* bias1 = (float*)take((size_t)G_ * 4);
  float* bias2 = (float*)take((size_t)G_ * 4);
  float* xp1   = (float*)take((size_t)S_ * B_ * G_ * 4);
  float* xp2   = (float*)take((size_t)S_ * B_ * G_ * 4);
  float* hs1   = (float*)take((size_t)B_ * S_ * E_ * 4);
  float* cs1   = (float*)take((size_t)B_ * S_ * E_ * 4);
  float* out2  = (float*)take((size_t)B_ * S_ * E_ * 4);
  float* baseb = (float*)take((size_t)B_ * S_ * E_ * 4);
  float* rnbuf = (float*)take((size_t)B_ * E_ * 4);

  k_cvt_pad<<<32, 256, 0, stream>>>(Wih1, wih1b, G_, D_, 64);
  k_cvt_pad<<<32, 256, 0, stream>>>(Wih2, wih2b, G_, D_, 64);
  k_cvt_pad<<<32, 256, 0, stream>>>(Whh1, whh1b, G_, E_, 64);
  k_cvt_pad<<<32, 256, 0, stream>>>(Whh2, whh2b, G_, E_, 64);
  k_cvt_T<<<16, 256, 0, stream>>>(wy, wyT, E_, E_);
  k_cvt_T<<<16, 256, 0, stream>>>(wh, whT, E_, E_);
  k_bias<<<1, 256, 0, stream>>>(bih1, bhh1, bias1, G_);
  k_bias<<<1, 256, 0, stream>>>(bih2, bhh2, bias2, G_);

  k_xproj<<<dim3(192, 2), 256, 0, stream>>>(sentence1, sentence2, emb,
                                            wih1b, wih2b, bias1, bias2, xp1, xp2);

  k_lstm<<<4, 256, 0, stream>>>(xp1, whh1b, hs1, cs1, nullptr, nullptr, nullptr);
  k_lstm<<<4, 256, 0, stream>>>(xp2, whh2b, out2, nullptr, s1_len, hs1, cs1);

  k_base<<<192, 256, 0, stream>>>(hs1, out2, wyT, whT, baseb);

  size_t attn_lds = (size_t)2 * S_ * AP_ * sizeof(float);   // ~195 KB dynamic LDS
  k_attn<<<64, 256, attn_lds, stream>>>(baseb, hs1, wr, wt, w, s1_s, s2_len, rnbuf);

  k_final<<<1, 256, 0, stream>>>(rnbuf, out2, s2_len, wp, wx, l1W, l1b, lW, lb, out);
}
